// fNIRS_TTT_LM_28930899706191
// MI455X (gfx1250) — compile-verified, matrix operand order checked by reading, not run on hardware
//
#include <hip/hip_runtime.h>
#include <math.h>

typedef __attribute__((ext_vector_type(16))) _Float16 v16h;
typedef __attribute__((ext_vector_type(8)))  _Float16 v8h;
typedef __attribute__((ext_vector_type(8)))  float    v8f;

#define WMMAF(a,b,c) __builtin_amdgcn_wmma_f32_16x16x32_f16(false,(a),false,(b),(short)0,(c),false,false)

// ---------------- constants ----------------
#define Bt   128
#define SP   256
#define DIMD 512
#define INCH 52
#define ICH  8
#define KT   30
#define H_CA 8
#define DH   64
#define NH   2
#define HD   128
#define Lttt 512
#define Sttt 256
#define MBS  16
#define NM   32

// ---------------- device helpers ----------------
__device__ inline float wsum(float v) {
#pragma unroll
  for (int m = 16; m >= 1; m >>= 1) v += __shfl_xor(v, m, 32);
  return v;
}
__device__ inline float wmaxf(float v) {
#pragma unroll
  for (int m = 16; m >= 1; m >>= 1) v = fmaxf(v, __shfl_xor(v, m, 32));
  return v;
}
__device__ inline float gelu_t(float x) {
  return 0.5f * x * (1.f + tanhf(0.79788456f * x * (1.f + 0.044715f * x * x)));
}
__device__ inline float gelu_b(float x) {
  float t = tanhf(0.79788456f * x * (1.f + 0.044715f * x * x));
  return 0.5f * x * ((1.f - t * t) * (0.79788456f + 0.1070322243f * x * x)) + 0.5f * (1.f + t);
}
__device__ inline v8f vzero() {
  v8f v;
#pragma unroll
  for (int i = 0; i < 8; ++i) v[i] = 0.f;
  return v;
}
// load 16 contiguous floats (16B-aligned) into a register array
__device__ inline void ld16(const float* __restrict__ src, float* r) {
  float4 a = *(const float4*)(src + 0);
  float4 b = *(const float4*)(src + 4);
  float4 c = *(const float4*)(src + 8);
  float4 d = *(const float4*)(src + 12);
  r[0]=a.x; r[1]=a.y; r[2]=a.z; r[3]=a.w;
  r[4]=b.x; r[5]=b.y; r[6]=b.z; r[7]=b.w;
  r[8]=c.x; r[9]=c.y; r[10]=c.z; r[11]=c.w;
  r[12]=d.x; r[13]=d.y; r[14]=d.z; r[15]=d.w;
}
// store 16 floats as f16 to LDS (two b128 stores)
__device__ inline void st16h(_Float16* dst, const float* r) {
  v8h h0, h1;
#pragma unroll
  for (int i = 0; i < 8; ++i) { h0[i] = (_Float16)r[i]; h1[i] = (_Float16)r[8 + i]; }
  *(v8h*)dst = h0;
  *(v8h*)(dst + 8) = h1;
}
// convert 8 contiguous f32 -> 8 f16 in LDS
__device__ inline void cv8(_Float16* dst, const float* __restrict__ src) {
  float4 a = *(const float4*)(src + 0);
  float4 b = *(const float4*)(src + 4);
  v8h h;
  h[0]=(_Float16)a.x; h[1]=(_Float16)a.y; h[2]=(_Float16)a.z; h[3]=(_Float16)a.w;
  h[4]=(_Float16)b.x; h[5]=(_Float16)b.y; h[6]=(_Float16)b.z; h[7]=(_Float16)b.w;
  *(v8h*)dst = h;
}
// A-fragment (16xK f16, row-major, ld = ldk) for v_wmma_f32_16x16x32_f16.
// lane<16: row=lane, k = kc+{0..7,16..23}; lane>=16: row=lane-16, k = kc+{8..15,24..31}
__device__ inline v16h ldsfragA(const _Float16* p, int ldk, int kc) {
  int lane = threadIdx.x & 31;
  int row = lane & 15;
  int kb = kc + ((lane & 16) ? 8 : 0);
  union { v16h v; v8h h[2]; } u;
  u.h[0] = *(const v8h*)(p + row * ldk + kb);
  u.h[1] = *(const v8h*)(p + row * ldk + kb + 16);
  return u.v;
}
// B-fragment from f32 global, K x N row-major (col n = n0+lane%16) -- strided
__device__ inline v16h gfragB_KN(const float* p, int ldn, int kc, int n0) {
  int lane = threadIdx.x & 31;
  int n = n0 + (lane & 15);
  int kb = kc + ((lane & 16) ? 8 : 0);
  v16h r;
#pragma unroll
  for (int j = 0; j < 16; ++j) {
    int k = kb + (j < 8 ? j : j + 8);
    r[j] = (_Float16)p[(size_t)k * ldn + n];
  }
  return r;
}
// B-fragment from f32 global, N x K row-major (A*B^T) -- vectorized contiguous
__device__ inline v16h gfragB_NK(const float* p, int ldk, int n0, int kc) {
  int lane = threadIdx.x & 31;
  int n = n0 + (lane & 15);
  int kb = kc + ((lane & 16) ? 8 : 0);
  const float* base = p + (size_t)n * ldk + kb;
  float4 a0 = *(const float4*)(base + 0);
  float4 a1 = *(const float4*)(base + 4);
  float4 b0 = *(const float4*)(base + 16);
  float4 b1 = *(const float4*)(base + 20);
  v16h r;
  r[0]=(_Float16)a0.x; r[1]=(_Float16)a0.y; r[2]=(_Float16)a0.z; r[3]=(_Float16)a0.w;
  r[4]=(_Float16)a1.x; r[5]=(_Float16)a1.y; r[6]=(_Float16)a1.z; r[7]=(_Float16)a1.w;
  r[8]=(_Float16)b0.x; r[9]=(_Float16)b0.y; r[10]=(_Float16)b0.z; r[11]=(_Float16)b0.w;
  r[12]=(_Float16)b1.x; r[13]=(_Float16)b1.y; r[14]=(_Float16)b1.z; r[15]=(_Float16)b1.w;
  return r;
}
// B-fragment from f16 LDS, K x N row-major, real K=16 padded to 32 with zeros
__device__ inline v16h ldsfragB_K16(const _Float16* p, int ldn) {
  int lane = threadIdx.x & 31;
  int nn = lane & 15;
  int kb = (lane & 16) ? 8 : 0;
  v16h r;
#pragma unroll
  for (int j = 0; j < 16; ++j) {
    int k = kb + (j < 8 ? j : j + 8);
    r[j] = (k < 16) ? p[k * ldn + nn] : (_Float16)0.f;
  }
  return r;
}

// ---------------- generic batched WMMA GEMM (requires M%64==0, N%64==0, K%32==0) ----------------
// C[z] = act(alpha * A[z] (MxK) * op(B[z]) + bias), opB=0: B is NxK (A*B^T), opB=1: B is KxN
__global__ void __launch_bounds__(128)
gemm_kernel(const float* __restrict__ A, const float* __restrict__ Bw,
            float* __restrict__ C, const float* __restrict__ bias,
            int M, int N, int K, int lda, int ldb, int ldc,
            float alpha, int act, int opB, int bdiv,
            long sA1, long sA2, long sB1, long sB2, long sC1, long sC2) {
  __shared__ __align__(16) _Float16 As[2 * 64 * 40];
  __shared__ __align__(16) _Float16 Bs[2 * 64 * 40];
  int tid = threadIdx.x;
  int z = blockIdx.z;
  long offA = (long)(z / bdiv) * sA1 + (long)(z % bdiv) * sA2;
  long offB = (long)(z / bdiv) * sB1 + (long)(z % bdiv) * sB2;
  long offC = (long)(z / bdiv) * sC1 + (long)(z % bdiv) * sC2;
  int m0 = blockIdx.y * 64, n0 = blockIdx.x * 64;
  int wv = tid >> 5, lane = tid & 31;
  int wr = (wv >> 1) * 32, wc = (wv & 1) * 32;

  // staging coordinates: thread t moves 16 floats (half row) of A and of B
  int arow = tid >> 1, ahalf = tid & 1;
  const float* Abase = A + offA + (size_t)(m0 + arow) * lda + ahalf * 16;
  const float* Bbase = Bw + offB + (size_t)(n0 + arow) * ldb + ahalf * 16; // opB==0
  int bq = tid & 15, bk4 = (tid >> 4) * 4;                                // opB==1

  float rA[16], rB[16];
  auto loadAB = [&](int kt) {
    ld16(Abase + kt, rA);
    if (opB == 0) {
      ld16(Bbase + kt, rB);
    } else {
#pragma unroll
      for (int kk = 0; kk < 4; ++kk) {
        float4 v = *(const float4*)(Bw + offB + (size_t)(kt + bk4 + kk) * ldb + n0 + bq * 4);
        rB[kk * 4 + 0] = v.x; rB[kk * 4 + 1] = v.y; rB[kk * 4 + 2] = v.z; rB[kk * 4 + 3] = v.w;
      }
    }
  };
  auto storeAB = [&](int p) {
    st16h(&As[p * 2560 + arow * 40 + ahalf * 16], rA);
    if (opB == 0) {
      st16h(&Bs[p * 2560 + arow * 40 + ahalf * 16], rB);
    } else {
#pragma unroll
      for (int kk = 0; kk < 4; ++kk)
#pragma unroll
        for (int j = 0; j < 4; ++j)
          Bs[p * 2560 + (bq * 4 + j) * 40 + bk4 + kk] = (_Float16)rB[kk * 4 + j];
    }
  };

  v8f acc[2][2];
  acc[0][0] = vzero(); acc[0][1] = vzero(); acc[1][0] = vzero(); acc[1][1] = vzero();

  loadAB(0);
  int p = 0;
  for (int kt = 0; kt < K; kt += 32) {
    storeAB(p);
    __syncthreads();
    if (kt + 64 < K) {
      __builtin_prefetch(Abase + kt + 64, 0, 1);
      if (opB == 0) __builtin_prefetch(Bbase + kt + 64, 0, 1);
    }
    if (kt + 32 < K) loadAB(kt + 32);
    const _Float16* Ap = As + p * 2560;
    const _Float16* Bp = Bs + p * 2560;
    v16h a0 = ldsfragA(Ap + (wr + 0) * 40, 40, 0);
    v16h a1 = ldsfragA(Ap + (wr + 16) * 40, 40, 0);
    v16h b0 = ldsfragA(Bp + (wc + 0) * 40, 40, 0);
    v16h b1 = ldsfragA(Bp + (wc + 16) * 40, 40, 0);
    acc[0][0] = WMMAF(a0, b0, acc[0][0]);
    acc[0][1] = WMMAF(a0, b1, acc[0][1]);
    acc[1][0] = WMMAF(a1, b0, acc[1][0]);
    acc[1][1] = WMMAF(a1, b1, acc[1][1]);
    p ^= 1;
  }
  int col = lane & 15, rb = (lane & 16) ? 8 : 0;
#pragma unroll
  for (int ti = 0; ti < 2; ++ti)
#pragma unroll
    for (int tj = 0; tj < 2; ++tj)
#pragma unroll
      for (int i = 0; i < 8; ++i) {
        int r = m0 + wr + ti * 16 + rb + i;
        int c = n0 + wc + tj * 16 + col;
        float v = acc[ti][tj][i] * alpha;
        if (bias) v += bias[c];
        if (act) v = fmaxf(v, 0.f);
        C[offC + (long)r * ldc + c] = v;
      }
}

// ---------------- conv patch embed ----------------
__global__ void conv_kernel(const float* __restrict__ img, const float* __restrict__ cw,
                            const float* __restrict__ cb, float* __restrict__ patches,
                            int KC, int padH, size_t total) {
  size_t id = (size_t)blockIdx.x * 256 + threadIdx.x;
  if (id >= total) return;
  int h = (int)(id % INCH); size_t r = id / INCH;
  int c = (int)(r % ICH); r /= ICH;
  int w = (int)(r % SP); int b = (int)(r / SP);
  float acc = cb[c];
  for (int ci = 0; ci < 2; ++ci)
    for (int kh = 0; kh < KC; ++kh) {
      int ih = h + kh - padH;
      if (ih < 0 || ih >= INCH) continue;
      for (int kw = 0; kw < KT; ++kw) {
        int iw = w + kw - 14;
        if (iw < 0 || iw >= SP) continue;
        acc += img[(((size_t)b * 2 + ci) * INCH + ih) * SP + iw] *
               cw[((c * 2 + ci) * KC + kh) * KT + kw];
      }
    }
  patches[((size_t)b * SP + w) * 416 + c * INCH + h] = acc;
}

// ---------------- layernorm over D=512 (+optional residual, +optional pos) ----------------
__global__ void ln512_kernel(const float* __restrict__ in, const float* __restrict__ res,
                             const float* __restrict__ g, const float* __restrict__ bb,
                             const float* __restrict__ pos, float* __restrict__ out,
                             int rows, float eps) {
  int wv = threadIdx.x >> 5, lane = threadIdx.x & 31;
  int row = blockIdx.x * 8 + wv;
  if (row >= rows) return;
  float x[16]; float s = 0.f;
#pragma unroll
  for (int e = 0; e < 16; ++e) {
    int d = lane + (e << 5);
    float v = in[(size_t)row * 512 + d];
    if (res) v += res[(size_t)row * 512 + d];
    x[e] = v; s += v;
  }
  float mu = wsum(s) * (1.f / 512.f);
  float vs = 0.f;
#pragma unroll
  for (int e = 0; e < 16; ++e) { float dd = x[e] - mu; vs += dd * dd; }
  float rstd = rsqrtf(wsum(vs) * (1.f / 512.f) + eps);
#pragma unroll
  for (int e = 0; e < 16; ++e) {
    int d = lane + (e << 5);
    float y = (x[e] - mu) * rstd * g[d] + bb[d];
    if (pos) y += pos[(size_t)(row % SP) * 512 + d];
    out[(size_t)row * 512 + d] = y;
  }
}

// ---------------- softmax over D=256 ----------------
__global__ void softmax_kernel(float* __restrict__ p, int rows) {
  int wv = threadIdx.x >> 5, lane = threadIdx.x & 31;
  int row = blockIdx.x * 8 + wv;
  if (row >= rows) return;
  float* r = p + (size_t)row * 256;
  float x[8]; float mx = -1e30f;
#pragma unroll
  for (int e = 0; e < 8; ++e) { x[e] = r[lane + (e << 5)]; mx = fmaxf(mx, x[e]); }
  mx = wmaxf(mx);
  float s = 0.f;
#pragma unroll
  for (int e = 0; e < 8; ++e) { x[e] = expf(x[e] - mx); s += x[e]; }
  float inv = 1.f / wsum(s);
#pragma unroll
  for (int e = 0; e < 8; ++e) r[lane + (e << 5)] = x[e] * inv;
}

// ---------------- transpose xres (B,SP,DIM) -> xt (B,DIM,SP) ----------------
__global__ void transpose_kernel(const float* __restrict__ xres, float* __restrict__ xt) {
  size_t id = (size_t)blockIdx.x * 256 + threadIdx.x;
  if (id >= (size_t)Bt * Lttt * Sttt) return;
  int s = (int)(id & 255); size_t bl = id >> 8;
  int l = (int)(bl & 511); int b = (int)(bl >> 9);
  xt[id] = xres[((size_t)b * SP + s) * DIMD + l];
}

// ---------------- rope + head split: (B,L,256) -> (B,NH,L,HD) ----------------
__global__ void rope_kernel(const float* __restrict__ in, float* __restrict__ out, int do_rope) {
  size_t id = (size_t)blockIdx.x * 256 + threadIdx.x;
  if (id >= (size_t)Bt * NH * Lttt * 64) return;
  int i = (int)(id & 63);
  int l = (int)((id >> 6) & 511);
  int h = (int)((id >> 15) & 1);
  int b = (int)(id >> 16);
  const float* src = in + ((size_t)b * Lttt + l) * 256 + h * HD;
  float x1 = src[i], x2 = src[64 + i], o1, o2;
  if (do_rope) {
    float inv = powf(10000.f, -((float)(2 * i)) * (1.f / 128.f));
    float e = (float)l * inv, c = cosf(e), sn = sinf(e);
    o1 = x1 * c - x2 * sn;
    o2 = x2 * c + x1 * sn;
  } else { o1 = x1; o2 = x2; }
  float* dst = out + (((size_t)b * NH + h) * Lttt + l) * HD;
  dst[i] = o1; dst[64 + i] = o2;
}

// ---------------- lr sigmoid: sig[(b*2+h)*512+l] = sigmoid(xt[b,l,:].lrw[h]+lrb[h])/HD ----------------
__global__ void sig_kernel(const float* __restrict__ xt, const float* __restrict__ lrw,
                           const float* __restrict__ lrb, float* __restrict__ sig) {
  int wv = threadIdx.x >> 5, lane = threadIdx.x & 31;
  int row = blockIdx.x * 8 + wv;
  if (row >= Bt * NH * Lttt) return;
  int l = row & 511, ch = row >> 9, h = ch & 1, b = ch >> 1;
  const float* xr = xt + ((size_t)b * Lttt + l) * 256;
  float s = 0.f;
#pragma unroll
  for (int e = 0; e < 8; ++e) { int c = lane + (e << 5); s += xr[c] * lrw[h * 256 + c]; }
  s = wsum(s) + lrb[h];
  if (lane == 0) sig[row] = (1.f / (1.f + expf(-s))) * (1.f / 128.f);
}

// ---------------- broadcast init ----------------
__global__ void bcast_kernel(float* __restrict__ dst, const float* __restrict__ src,
                             size_t total, int mod) {
  size_t i = (size_t)blockIdx.x * 256 + threadIdx.x;
  if (i < total) dst[i] = src[i % mod];
}
// transposing broadcast: per chain ch (h=ch&1), dst (R x C) row-major = src[h] (C x R)^T
__global__ void bcastT_kernel(float* __restrict__ dst, const float* __restrict__ src,
                              int R, int C, size_t total) {
  size_t i = (size_t)blockIdx.x * 256 + threadIdx.x;
  if (i >= total) return;
  int mat = R * C;
  int rc = (int)(i % (size_t)mat);
  int ch = (int)(i / (size_t)mat);
  int h = ch & 1;
  int r = rc / C, c = rc % C;
  dst[i] = src[(size_t)h * mat + (size_t)c * R + r];
}

// ---------------- fused TTT chain: one workgroup per (b,h) chain, 32 steps ----------------
// Fast weights stored transposed: W1T (512x128) with W1T[o][d], W2T (128x512) with W2T[d][o]
__global__ void __launch_bounds__(256)
ttt_chain_kernel(const float* __restrict__ XQ, const float* __restrict__ XK,
                 const float* __restrict__ XV, const float* __restrict__ sig,
                 const float* __restrict__ tok, const float* __restrict__ lng,
                 const float* __restrict__ lnb,
                 float* __restrict__ W1, float* __restrict__ W2,
                 float* __restrict__ b1, float* __restrict__ b2,
                 float* __restrict__ outp) {
  __shared__ __align__(16) _Float16 hXQ[16 * 128], hXK[16 * 128], hXV[16 * 128];
  __shared__ __align__(16) _Float16 hZ1[16 * 512];   // Z1 -> X2b (reused)
  __shared__ __align__(16) _Float16 hX2[16 * 512];
  __shared__ __align__(16) _Float16 hgZ1[16 * 512];
  __shared__ __align__(16) _Float16 hgZ2[16 * 128];
  __shared__ __align__(16) _Float16 hM[16 * 32];
  __shared__ float fZ2[16 * 128];
  __shared__ float fZ2b[16 * 128];
  __shared__ float fEta[256];

  int ch = blockIdx.x;
  int b = ch >> 1, h = ch & 1;
  int tid = threadIdx.x, lane = tid & 31, wv = tid >> 5;
  const float* xqg = XQ + (size_t)ch * Lttt * HD;
  const float* xkg = XK + (size_t)ch * Lttt * HD;
  const float* xvg = XV + (size_t)ch * Lttt * HD;
  float* W1g = W1 + (size_t)ch * 512 * 128;  // W1T
  float* W2g = W2 + (size_t)ch * 128 * 512;  // W2T
  float* b1g = b1 + (size_t)ch * 512;
  float* b2g = b2 + (size_t)ch * 128;
  const float* lg = lng + h * 128;
  const float* lb = lnb + h * 128;
  int col = lane & 15, rb = (lane & 16) ? 8 : 0;

  for (int n = 0; n < NM; ++n) {
    int l0 = n * MBS;
    __syncthreads();
    // stage xq/xk/xv (vectorized) + eta
    {
      int i0 = tid * 8;
      cv8(&hXQ[i0], xqg + (size_t)l0 * 128 + i0);
      cv8(&hXK[i0], xkg + (size_t)l0 * 128 + i0);
      cv8(&hXV[i0], xvg + (size_t)l0 * 128 + i0);
      int i = tid >> 4, j = tid & 15;
      float ti = fmaxf(1.f / (float)(i + 1) + tok[i], 0.f);
      fEta[tid] = ti * sig[(size_t)ch * Lttt + l0 + j];
    }
    __syncthreads();

    // step1: Z1 = xk@W1+b1 ; X2 = gelu(Z1) ; accQ = xq@W1 kept in VGPRs
    v8f accQ[4];
#pragma unroll
    for (int tt = 0; tt < 4; ++tt) {
      int nn0 = (wv + tt * 8) * 16;
      v8f aK = vzero(), aQ = vzero();
#pragma unroll
      for (int kc = 0; kc < 128; kc += 32) {
        v16h fa = ldsfragA(hXK, 128, kc);
        v16h fq = ldsfragA(hXQ, 128, kc);
        v16h fb = gfragB_NK(W1g, 128, nn0, kc);   // W1T rows o contiguous over d
        aK = WMMAF(fa, fb, aK);
        aQ = WMMAF(fq, fb, aQ);
      }
      accQ[tt] = aQ;
      int c = nn0 + col;
      float bbv = b1g[c];
#pragma unroll
      for (int i = 0; i < 8; ++i) {
        float z = aK[i] + bbv;
        hZ1[(rb + i) * 512 + c] = (_Float16)z;
        hX2[(rb + i) * 512 + c] = (_Float16)gelu_t(z);
      }
    }
    __syncthreads();

    // step2: Z2 = X2@W2 + b2
    {
      int nn0 = wv * 16;
      v8f a = vzero();
#pragma unroll
      for (int kc = 0; kc < 512; kc += 32) {
        v16h fa = ldsfragA(hX2, 512, kc);
        v16h fb = gfragB_NK(W2g, 512, nn0, kc);   // W2T rows d contiguous over o
        a = WMMAF(fa, fb, a);
      }
      int c = nn0 + col;
      float bbv = b2g[c];
#pragma unroll
      for (int i = 0; i < 8; ++i) fZ2[(rb + i) * 128 + c] = a[i] + bbv;
    }
    __syncthreads();

    // step3: gZ2 = ln_l2_bwd(Z2, xv-xk, lw, lb)  (per row, D=128, eps 1e-6)
#pragma unroll
    for (int rr = 0; rr < 2; ++rr) {
      int r = wv * 2 + rr;
      float x[4]; float s = 0.f;
#pragma unroll
      for (int e = 0; e < 4; ++e) { int d = lane + (e << 5); x[e] = fZ2[r * 128 + d]; s += x[e]; }
      float mu = wsum(s) * (1.f / 128.f);
      float vs = 0.f;
#pragma unroll
      for (int e = 0; e < 4; ++e) { float dd = x[e] - mu; vs += dd * dd; }
      float rstd = rsqrtf(wsum(vs) * (1.f / 128.f) + 1e-6f);
      float gy[4], xh[4]; float s1 = 0.f, s2 = 0.f;
#pragma unroll
      for (int e = 0; e < 4; ++e) {
        int d = lane + (e << 5);
        xh[e] = (x[e] - mu) * rstd;
        float tgt = (float)hXV[r * 128 + d] - (float)hXK[r * 128 + d];
        gy[e] = (lg[d] * xh[e] + lb[d] - tgt) * lg[d];
        s1 += gy[e]; s2 += gy[e] * xh[e];
      }
      s1 = wsum(s1); s2 = wsum(s2);
#pragma unroll
      for (int e = 0; e < 4; ++e) {
        int d = lane + (e << 5);
        float g2 = (128.f * gy[e] - s1 - xh[e] * s2) * (rstd * (1.f / 128.f));
        hgZ2[r * 128 + d] = (_Float16)g2;
      }
    }
    __syncthreads();

    // step4: gZ1 = (gZ2 @ W2^T) * gelu_bwd(Z1)   (W2T used as KxN, strided)
#pragma unroll
    for (int tt = 0; tt < 4; ++tt) {
      int nn0 = (wv + tt * 8) * 16;
      v8f a = vzero();
#pragma unroll
      for (int kc = 0; kc < 128; kc += 32) {
        v16h fa = ldsfragA(hgZ2, 128, kc);
        v16h fb = gfragB_KN(W2g, 512, kc, nn0);
        a = WMMAF(fa, fb, a);
      }
      int c = nn0 + col;
#pragma unroll
      for (int i = 0; i < 8; ++i) {
        float z1 = (float)hZ1[(rb + i) * 512 + c];
        hgZ1[(rb + i) * 512 + c] = (_Float16)(a[i] * gelu_b(z1));
      }
    }
    __syncthreads();

    // step5: M1 = eta o (tril(xq xk^T) + tril)
    {
      int i = tid >> 4, j = tid & 15;
      float sdot = 0.f;
      for (int d = 0; d < 128; ++d) sdot += (float)hXQ[i * 128 + d] * (float)hXK[j * 128 + d];
      hM[i * 32 + j] = (_Float16)((i >= j) ? fEta[i * 16 + j] * (sdot + 1.f) : 0.f);
      hM[i * 32 + 16 + j] = (_Float16)0.f;
    }
    __syncthreads();

    // step6: Z1b = xq@W1 - M1@gZ1 + b1 ; X2b = gelu(Z1b) -> hZ1 (reuse)
#pragma unroll
    for (int tt = 0; tt < 4; ++tt) {
      int nn0 = (wv + tt * 8) * 16;
      v16h fa = ldsfragA(hM, 32, 0);
      v16h fb = ldsfragB_K16(hgZ1 + nn0, 512);
      v8f a = WMMAF(fa, fb, vzero());
      int c = nn0 + col;
      float bbv = b1g[c];
#pragma unroll
      for (int i = 0; i < 8; ++i) {
        float z = accQ[tt][i] - a[i] + bbv;
        hZ1[(rb + i) * 512 + c] = (_Float16)gelu_t(z);
      }
    }
    __syncthreads();

    // step7: M2 = eta o (tril(X2b X2^T) + tril)
    {
      int i = tid >> 4, j = tid & 15;
      float sdot = 0.f;
      for (int d = 0; d < 512; ++d) sdot += (float)hZ1[i * 512 + d] * (float)hX2[j * 512 + d];
      hM[i * 32 + j] = (_Float16)((i >= j) ? fEta[i * 16 + j] * (sdot + 1.f) : 0.f);
      hM[i * 32 + 16 + j] = (_Float16)0.f;
    }
    __syncthreads();

    // step8: Z2b = X2b@W2 - M2@gZ2 + b2
    {
      int nn0 = wv * 16;
      v8f a = vzero();
#pragma unroll
      for (int kc = 0; kc < 512; kc += 32) {
        v16h fa = ldsfragA(hZ1, 512, kc);
        v16h fb = gfragB_NK(W2g, 512, nn0, kc);
        a = WMMAF(fa, fb, a);
      }
      v16h ma = ldsfragA(hM, 32, 0);
      v16h mb = ldsfragB_K16(hgZ2 + nn0, 128);
      v8f m = WMMAF(ma, mb, vzero());
      int c = nn0 + col;
      float bbv = b2g[c];
#pragma unroll
      for (int i = 0; i < 8; ++i) fZ2b[(rb + i) * 128 + c] = a[i] - m[i] + bbv;
    }
    __syncthreads();

    // step9: out = xq + ln_fwd(Z2b) (eps 1e-6)
#pragma unroll
    for (int rr = 0; rr < 2; ++rr) {
      int r = wv * 2 + rr;
      float x[4]; float s = 0.f;
#pragma unroll
      for (int e = 0; e < 4; ++e) { int d = lane + (e << 5); x[e] = fZ2b[r * 128 + d]; s += x[e]; }
      float mu = wsum(s) * (1.f / 128.f);
      float vs = 0.f;
#pragma unroll
      for (int e = 0; e < 4; ++e) { float dd = x[e] - mu; vs += dd * dd; }
      float rstd = rsqrtf(wsum(vs) * (1.f / 128.f) + 1e-6f);
#pragma unroll
      for (int e = 0; e < 4; ++e) {
        int d = lane + (e << 5);
        float y = lg[d] * (x[e] - mu) * rstd + lb[d] + (float)hXQ[r * 128 + d];
        outp[((size_t)b * Lttt + l0 + r) * 256 + h * HD + d] = y;
      }
    }

    // step10: fast-weight updates (rank-16), transposed layouts
    for (int idx = tid; idx < 512 * 128; idx += 256) {       // W1T[o][d]
      int o = idx >> 7, d = idx & 127;
      float acc = 0.f;
#pragma unroll
      for (int j = 0; j < 16; ++j)
        acc += fEta[240 + j] * (float)hXK[j * 128 + d] * (float)hgZ1[j * 512 + o];
      W1g[idx] -= acc;
    }
    for (int idx = tid; idx < 128 * 512; idx += 256) {       // W2T[d][o]
      int d = idx >> 9, o = idx & 511;
      float acc = 0.f;
#pragma unroll
      for (int j = 0; j < 16; ++j)
        acc += fEta[240 + j] * (float)hX2[j * 512 + o] * (float)hgZ2[j * 128 + d];
      W2g[idx] -= acc;
    }
    for (int o = tid; o < 512; o += 256) {
      float acc = 0.f;
#pragma unroll
      for (int j = 0; j < 16; ++j) acc += fEta[240 + j] * (float)hgZ1[j * 512 + o];
      b1g[o] -= acc;
    }
    for (int d = tid; d < 128; d += 256) {
      float acc = 0.f;
#pragma unroll
      for (int j = 0; j < 16; ++j) acc += fEta[240 + j] * (float)hgZ2[j * 128 + d];
      b2g[d] -= acc;
    }
    __threadfence();
  }
}

// ---------------- post-TTT: cls[b*512+l] = LN256(tout[b,l,:]).wo0 + xres[b,0,l] ----------------
__global__ void tout_head_kernel(const float* __restrict__ tout, const float* __restrict__ pg,
                                 const float* __restrict__ pb, const float* __restrict__ wo,
                                 const float* __restrict__ xres, float* __restrict__ cls) {
  int wv = threadIdx.x >> 5, lane = threadIdx.x & 31;
  int row = blockIdx.x * 8 + wv;
  if (row >= Bt * Lttt) return;
  const float* r = tout + (size_t)row * 256;
  float x[8]; float s = 0.f;
#pragma unroll
  for (int e = 0; e < 8; ++e) { x[e] = r[lane + (e << 5)]; s += x[e]; }
  float mu = wsum(s) * (1.f / 256.f);
  float vs = 0.f;
#pragma unroll
  for (int e = 0; e < 8; ++e) { float dd = x[e] - mu; vs += dd * dd; }
  float rstd = rsqrtf(wsum(vs) * (1.f / 256.f) + 1e-5f);
  float dot = 0.f;
#pragma unroll
  for (int e = 0; e < 8; ++e) {
    int d = lane + (e << 5);
    dot += ((x[e] - mu) * rstd * pg[d] + pb[d]) * wo[d];
  }
  dot = wsum(dot);
  if (lane == 0) {
    int b = row >> 9, l = row & 511;
    cls[row] = dot + xres[(size_t)b * SP * DIMD + l];
  }
}

// ---------------- classifier head ----------------
__global__ void head_kernel(const float* __restrict__ cls, const float* __restrict__ g,
                            const float* __restrict__ bb, const float* __restrict__ w,
                            const float* __restrict__ hb, float* __restrict__ out) {
  int wv = threadIdx.x >> 5, lane = threadIdx.x & 31;
  int row = blockIdx.x * 8 + wv;
  if (row >= Bt) return;
  float x[16]; float s = 0.f;
#pragma unroll
  for (int e = 0; e < 16; ++e) { x[e] = cls[(size_t)row * 512 + lane + (e << 5)]; s += x[e]; }
  float mu = wsum(s) * (1.f / 512.f);
  float vs = 0.f;
#pragma unroll
  for (int e = 0; e < 16; ++e) { float dd = x[e] - mu; vs += dd * dd; }
  float rstd = rsqrtf(wsum(vs) * (1.f / 512.f) + 1e-5f);
  float y[16];
#pragma unroll
  for (int e = 0; e < 16; ++e) {
    int d = lane + (e << 5);
    y[e] = (x[e] - mu) * rstd * g[d] + bb[d];
  }
  for (int c = 0; c < 3; ++c) {
    float dot = 0.f;
#pragma unroll
    for (int e = 0; e < 16; ++e) dot += y[e] * w[c * 512 + lane + (e << 5)];
    dot = wsum(dot);
    if (lane == 0) out[row * 3 + c] = dot + hb[c];
  }
}

// ---------------- host orchestration ----------------
static inline void gemm_launch(hipStream_t st, const float* A, const float* Bw, float* C,
                               const float* bias, int M, int N, int K, int lda, int ldb, int ldc,
                               float alpha, int act, int opB, int batch, int bdiv,
                               long sA1, long sA2, long sB1, long sB2, long sC1, long sC2) {
  dim3 g((N + 63) / 64, (M + 63) / 64, batch);
  gemm_kernel<<<g, 128, 0, st>>>(A, Bw, C, bias, M, N, K, lda, ldb, ldc,
                                 alpha, act, opB, bdiv, sA1, sA2, sB1, sB2, sC1, sC2);
}

extern "C" void kernel_launch(void* const* d_in, const int* in_sizes, int n_in,
                              void* d_out, int out_size, void* d_ws, size_t ws_size,
                              hipStream_t stream) {
  (void)in_sizes; (void)n_in; (void)out_size; (void)ws_size;
  const float* img    = (const float*)d_in[0];
  const float* pe_cw  = (const float*)d_in[1];
  const float* pe_cb  = (const float*)d_in[2];
  const float* pe_lw  = (const float*)d_in[3];
  const float* pe_lb  = (const float*)d_in[4];
  const float* pe_g   = (const float*)d_in[5];
  const float* pe_b   = (const float*)d_in[6];
  const float* ce_cw  = (const float*)d_in[7];
  const float* ce_cb  = (const float*)d_in[8];
  const float* ce_lw  = (const float*)d_in[9];
  const float* ce_lb  = (const float*)d_in[10];
  const float* ce_g   = (const float*)d_in[11];
  const float* ce_b   = (const float*)d_in[12];
  const float* pos_p  = (const float*)d_in[13];
  const float* pos_c  = (const float*)d_in[14];
  const float* ca_wq  = (const float*)d_in[15];
  const float* ca_bq  = (const float*)d_in[16];
  const float* ca_wk  = (const float*)d_in[17];
  const float* ca_bk  = (const float*)d_in[18];
  const float* ca_wv  = (const float*)d_in[19];
  const float* ca_bv  = (const float*)d_in[20];
  const float* ca_wo  = (const float*)d_in[21];
  const float* ca_bo  = (const float*)d_in[22];
  const float* ca_g1  = (const float*)d_in[23];
  const float* ca_b1  = (const float*)d_in[24];
  const float* ca_fw1 = (const float*)d_in[25];
  const float* ca_fb1 = (const float*)d_in[26];
  const float* ca_fw2 = (const float*)d_in[27];
  const float* ca_fb2 = (const float*)d_in[28];
  const float* ca_g2  = (const float*)d_in[29];
  const float* ca_b2  = (const float*)d_in[30];
  const float* pre_g  = (const float*)d_in[31];
  const float* pre_b  = (const float*)d_in[32];
  const float* t_wq   = (const float*)d_in[33];
  const float* t_wk   = (const float*)d_in[34];
  const float* t_wv   = (const float*)d_in[35];
  const float* t_wo   = (const float*)d_in[36];
  const float* t_W1   = (const float*)d_in[37];
  const float* t_b1   = (const float*)d_in[38];
  const float* t_W2   = (const float*)d_in[39];
  const float* t_b2   = (const float*)d_in[40];
  const float* t_lng  = (const float*)d_in[41];
  const float* t_lnb  = (const float*)d_in[42];
  const float* t_lrw  = (const float*)d_in[43];
  const float* t_lrb  = (const float*)d_in[44];
  const float* t_tok  = (const float*)d_in[45];
  const float* t_postg= (const float*)d_in[46];
  const float* t_postb= (const float*)d_in[47];
  const float* h_g    = (const float*)d_in[48];
  const float* h_b    = (const float*)d_in[49];
  const float* h_w    = (const float*)d_in[50];
  const float* h_bb   = (const float*)d_in[51];
  float* out = (float*)d_out;

  float* w = (float*)d_ws;
  size_t off = 0;
  auto alloc = [&](size_t n) { float* p = w + off; off += (n + 63) & ~(size_t)63; return p; };
  const size_t TOK = (size_t)Bt * SP * DIMD;            // 16.78M
  float* patches = alloc((size_t)Bt * SP * 416);
  float* xP   = alloc(TOK);
  float* xC   = alloc(TOK);
  float* bufQ = alloc(TOK);
  float* bufK = alloc(TOK);
  float* bufV = alloc(TOK);
  float* bufO = alloc(TOK);
  float* bufT1= alloc(TOK);
  float* bufT2= alloc(TOK);
  float* bufS = alloc((size_t)Bt * SP * 2048);          // 67.1M (scores / ffn hidden)
  float* xres = alloc(TOK);
  float* xt   = alloc(TOK);
  float* XQr  = alloc(TOK);
  float* XKr  = alloc(TOK);
  float* XVr  = alloc(TOK);
  float* sigb = alloc((size_t)Bt * NH * Lttt);
  float* chW1 = alloc((size_t)Bt * NH * 512 * 128);     // W1T per chain
  float* chW2 = alloc((size_t)Bt * NH * 128 * 512);     // W2T per chain
  float* chb1 = alloc((size_t)Bt * NH * 512);
  float* chb2 = alloc((size_t)Bt * NH * 128);
  float* tout = alloc(TOK);
  float* cls  = alloc((size_t)Bt * 512);

  const int ROWS = Bt * SP;                              // 32768
  const size_t convTot = (size_t)Bt * SP * ICH * INCH;   // 13.6M

  // ---- patch embeds ----
  conv_kernel<<<(convTot + 255) / 256, 256, 0, stream>>>(img, pe_cw, pe_cb, patches, 5, 2, convTot);
  gemm_launch(stream, patches, pe_lw, xP, pe_lb, ROWS, 512, 416, 416, 416, 512, 1.f, 0, 0, 1, 1, 0,0,0,0,0,0);
  ln512_kernel<<<ROWS / 8, 256, 0, stream>>>(xP, nullptr, pe_g, pe_b, pos_p, xP, ROWS, 1e-5f);
  conv_kernel<<<(convTot + 255) / 256, 256, 0, stream>>>(img, ce_cw, ce_cb, patches, 1, 0, convTot);
  gemm_launch(stream, patches, ce_lw, xC, ce_lb, ROWS, 512, 416, 416, 416, 512, 1.f, 0, 0, 1, 1, 0,0,0,0,0,0);
  ln512_kernel<<<ROWS / 8, 256, 0, stream>>>(xC, nullptr, ce_g, ce_b, pos_c, xC, ROWS, 1e-5f);

  // ---- cross attention ----
  gemm_launch(stream, xP, ca_wq, bufQ, ca_bq, ROWS, 512, 512, 512, 512, 512, 1.f, 0, 0, 1, 1, 0,0,0,0,0,0);
  gemm_launch(stream, xC, ca_wk, bufK, ca_bk, ROWS, 512, 512, 512, 512, 512, 1.f, 0, 0, 1, 1, 0,0,0,0,0,0);
  gemm_launch(stream, xC, ca_wv, bufV, ca_bv, ROWS, 512, 512, 512, 512, 512, 1.f, 0, 0, 1, 1, 0,0,0,0,0,0);
  // scores[b,h] = Q_bh K_bh^T / 8   (batch = B*H)
  gemm_launch(stream, bufQ, bufK, bufS, nullptr, 256, 256, 64, 512, 512, 256, 0.125f, 0, 0,
              Bt * H_CA, H_CA, 131072, 64, 131072, 64, 524288, 65536);
  softmax_kernel<<<(Bt * H_CA * 256) / 8, 256, 0, stream>>>(bufS, Bt * H_CA * 256);
  // O_bh = att V_bh   (B is KxN -> opB=1)
  gemm_launch(stream, bufS, bufV, bufO, nullptr, 256, 64, 256, 256, 512, 512, 1.f, 0, 1,
              Bt * H_CA, H_CA, 524288, 65536, 131072, 64, 131072, 64);
  gemm_launch(stream, bufO, ca_wo, bufT2, ca_bo, ROWS, 512, 512, 512, 512, 512, 1.f, 0, 0, 1, 1, 0,0,0,0,0,0);
  ln512_kernel<<<ROWS / 8, 256, 0, stream>>>(bufT2, xP, ca_g1, ca_b1, nullptr, bufT1, ROWS, 1e-5f);
  gemm_launch(stream, bufT1, ca_fw1, bufS, ca_fb1, ROWS, 2048, 512, 512, 512, 2048, 1.f, 1, 0, 1, 1, 0,0,0,0,0,0);
  gemm_launch(stream, bufS, ca_fw2, bufT2, ca_fb2, ROWS, 512, 2048, 2048, 2048, 512, 1.f, 0, 0, 1, 1, 0,0,0,0,0,0);
  ln512_kernel<<<ROWS / 8, 256, 0, stream>>>(bufT2, bufT1, ca_g2, ca_b2, nullptr, bufT2, ROWS, 1e-5f);
  ln512_kernel<<<ROWS / 8, 256, 0, stream>>>(bufT2, nullptr, pre_g, pre_b, nullptr, xres, ROWS, 1e-5f);

  // ---- TTT ----
  transpose_kernel<<<(TOK + 255) / 256, 256, 0, stream>>>(xres, xt);
  gemm_launch(stream, xt, t_wq, bufQ, nullptr, Bt * Lttt, 256, 256, 256, 256, 256, 1.f, 0, 0, 1, 1, 0,0,0,0,0,0);
  gemm_launch(stream, xt, t_wk, bufK, nullptr, Bt * Lttt, 256, 256, 256, 256, 256, 1.f, 0, 0, 1, 1, 0,0,0,0,0,0);
  gemm_launch(stream, xt, t_wv, bufV, nullptr, Bt * Lttt, 256, 256, 256, 256, 256, 1.f, 0, 0, 1, 1, 0,0,0,0,0,0);
  const size_t ropeTot = (size_t)Bt * NH * Lttt * 64;
  rope_kernel<<<(ropeTot + 255) / 256, 256, 0, stream>>>(bufQ, XQr, 1);
  rope_kernel<<<(ropeTot + 255) / 256, 256, 0, stream>>>(bufK, XKr, 1);
  rope_kernel<<<(ropeTot + 255) / 256, 256, 0, stream>>>(bufV, XVr, 0);
  sig_kernel<<<(Bt * NH * Lttt) / 8, 256, 0, stream>>>(xt, t_lrw, t_lrb, sigb);
  {
    size_t tW = (size_t)Bt * NH * 65536;
    bcastT_kernel<<<(tW + 255) / 256, 256, 0, stream>>>(chW1, t_W1, 512, 128, tW);  // W1T
    bcastT_kernel<<<(tW + 255) / 256, 256, 0, stream>>>(chW2, t_W2, 128, 512, tW);  // W2T
    size_t tb1 = (size_t)Bt * NH * 512, tb2 = (size_t)Bt * NH * 128;
    bcast_kernel<<<(tb1 + 255) / 256, 256, 0, stream>>>(chb1, t_b1, tb1, NH * 512);
    bcast_kernel<<<(tb2 + 255) / 256, 256, 0, stream>>>(chb2, t_b2, tb2, NH * 128);
  }
  ttt_chain_kernel<<<Bt * NH, 256, 0, stream>>>(XQr, XKr, XVr, sigb, t_tok, t_lng, t_lnb,
                                                chW1, chW2, chb1, chb2, tout);

  // ---- head (only column s=0 of the final transpose is consumed) ----
  tout_head_kernel<<<(Bt * Lttt) / 8, 256, 0, stream>>>(tout, t_postg, t_postb, t_wo, xres, cls);
  head_kernel<<<(Bt + 7) / 8, 256, 0, stream>>>(cls, h_g, h_b, h_w, h_bb, out);
}
